// VisualTextualRelationAwareModule_55465207661219
// MI455X (gfx1250) — matrix operand order repaired and checked by reading, not compiled, not run on hardware
//
#include <hip/hip_runtime.h>
#include <math.h>

// ---------------------------------------------------------------------------
// Fused VisualTextualRelationAware module for MI455X (gfx1250, wave32, WMMA).
//
// Layout convention: all per-batch activations kept as (HW x C) = (49 x 256)
// row-major tiles, padded to 64 rows, bf16 in LDS.  Every 1x1 conv becomes
//   out(64x256) = A(64x256) @ W^T  with  B[k][n] = W[n][k]  == W row-major,
// done with v_wmma_f32_16x16x32_bf16 (M-tiles x N-tiles x K-steps = 4x2x8 per
// wave, 8 waves cover N=256).
//
//  - non-temporal loads of x / stores of out (each is a 205MB single-use
//    stream; keeps the 512KB bf16 weights + language K/V resident in the
//    192MB L2 across all 4096 workgroups)
//  - language K/V + softmax bias staged into LDS via
//    GLOBAL_LOAD_ASYNC_TO_LDS_B128 issued at kernel entry, drained with
//    s_wait_asynccnt right before the attention stage (overlaps the first 2 GEMMs)
// ---------------------------------------------------------------------------

typedef __attribute__((ext_vector_type(8)))  float  v8f;
typedef __attribute__((ext_vector_type(8)))  __bf16 v8bf;
typedef __attribute__((ext_vector_type(16))) __bf16 v16bf;

// async-to-LDS builtin operand types (signature discovered from diagnostics:
// arg0 = AS1 pointer to int __vector(4), arg1 = LDS side, 2 imm ints)
typedef int v4i_ __attribute__((vector_size(16)));
typedef __attribute__((address_space(1))) v4i_* gv4p;
typedef __attribute__((address_space(3))) v4i_* lv4p;

#define HW    49
#define MPAD  64
#define CDIM  256
#define LC    768
#define NL    20
#define KH    128   // per-head key/value dim (256 / 2 heads)
#define EPSV  1e-5f

// klang (256*20 f32) | vlang (256*20 f32) | simbias (20 f32), contiguous
#define KVBYTES   (2 * CDIM * NL * 4 + NL * 4)          // 41040
#define KVCHUNKS  ((KVBYTES + 15) / 16)                 // 2565 x 16B

#if __has_builtin(__builtin_amdgcn_global_load_async_to_lds_b128)
#define HAVE_ASYNC_LDS 1
#else
#define HAVE_ASYNC_LDS 0
#endif

__device__ __forceinline__ float gelu_exact(float x) {
  return 0.5f * x * (1.0f + erff(x * 0.70710678118654752f));
}

__device__ __forceinline__ v16bf cat16(v8bf lo, v8bf hi) {
  v16bf r;
#pragma unroll
  for (int i = 0; i < 8; ++i) { r[i] = lo[i]; r[i + 8] = hi[i]; }
  return r;
}

// out[m][n] = act( sum_k A[m][k] * Wg[n*256+k] + bias[n] )
// A: MPADxCDIM bf16 in LDS (rows >= HW are zero).  Wg: bf16 row-major 256x256
// in global (L2-resident).  Result to smOut (bf16 LDS) or gOut (f32 global,
// rows m < HW only, non-temporal stores).
__device__ __forceinline__ void gemm_fused(const __bf16* __restrict__ smA,
                                           const __bf16* __restrict__ Wg,
                                           const float*  __restrict__ bias,
                                           int do_gelu,
                                           __bf16* smOut,
                                           float* __restrict__ gOut)
{
  const int lane  = threadIdx.x & 31;
  const int wave  = threadIdx.x >> 5;
  const int nlo   = lane & 15;
  const int hi    = lane >> 4;        // 0 or 1 (half-wave)
  const int nbase = wave * 32;        // 2 N-tiles per wave

  v8f acc[4][2];
#pragma unroll
  for (int mt = 0; mt < 4; ++mt)
#pragma unroll
    for (int nt = 0; nt < 2; ++nt) acc[mt][nt] = v8f{};

#pragma unroll
  for (int ks = 0; ks < 8; ++ks) {
    // A fragment (16x32 bf16): lanes 0-15 hold M=nlo, K = ks*32 + {0..7,16..23}
    //                          lanes 16-31 hold K = ks*32 + {8..15,24..31}
    const int kA = ks * 32 + hi * 8;
    v16bf afr[4];
#pragma unroll
    for (int mt = 0; mt < 4; ++mt) {
      const __bf16* pa = smA + (mt * 16 + nlo) * CDIM + kA;
      afr[mt] = cat16(*(const v8bf*)pa, *(const v8bf*)(pa + 16));
    }
    // B fragment (32x16 bf16): lane holds column n=nlo, K = ks*32 + hi*8 + {0..7,16..23}
    const int kB = ks * 32 + hi * 8;
#pragma unroll
    for (int nt = 0; nt < 2; ++nt) {
      const __bf16* pb = Wg + (size_t)(nbase + nt * 16 + nlo) * CDIM + kB;
      v16bf bfr = cat16(*(const v8bf*)pb, *(const v8bf*)(pb + 16));
#pragma unroll
      for (int mt = 0; mt < 4; ++mt)
        acc[mt][nt] = __builtin_amdgcn_wmma_f32_16x16x32_bf16(
            false, afr[mt], false, bfr, (short)0, acc[mt][nt], false, false);
    }
  }

  // Epilogue: C/D layout -> VGPR r holds M = mt*16 + hi*8 + r, N = nbase+nt*16+nlo
#pragma unroll
  for (int nt = 0; nt < 2; ++nt) {
    const int n  = nbase + nt * 16 + nlo;
    const float bn = bias[n];
#pragma unroll
    for (int mt = 0; mt < 4; ++mt) {
#pragma unroll
      for (int r = 0; r < 8; ++r) {
        float v = acc[mt][nt][r] + bn;
        if (do_gelu) v = gelu_exact(v);
        const int m = mt * 16 + hi * 8 + r;
        if (smOut) {
          smOut[m * CDIM + n] = (__bf16)v;
        } else if (m < HW) {
          __builtin_nontemporal_store(v, gOut + (size_t)m * CDIM + n);
        }
      }
    }
  }
}

// --- preprocessing: f32 weights -> bf16 copies in workspace --------------------
__global__ __launch_bounds__(256)
void convert_weights(const float* __restrict__ a, const float* __restrict__ b,
                     const float* __restrict__ c, const float* __restrict__ d,
                     __bf16* __restrict__ dst)
{
  const int i   = blockIdx.x * 256 + threadIdx.x;   // 0 .. 4*65536-1
  const int mat = i >> 16;
  const int off = i & 65535;
  const float* src = (mat == 0) ? a : (mat == 1) ? b : (mat == 2) ? c : d;
  dst[i] = (__bf16)src[off];
}

// --- preprocessing: language K/V (broadcast over all batches, so do it once) --
__global__ __launch_bounds__(256)
void build_kv(const float* __restrict__ l, const float* __restrict__ l_mask,
              const float* __restrict__ w_k, const float* __restrict__ b_k,
              const float* __restrict__ w_v, const float* __restrict__ b_v,
              float* __restrict__ klang, float* __restrict__ vlang,
              float* __restrict__ simbias)
{
  const int n = blockIdx.x;     // 0..19 (text position)
  const int t = threadIdx.x;    // 0..255 (output channel)
  const float mk = l_mask[n];
  float sk = b_k[t], sv = b_v[t];
  for (int c = 0; c < LC; ++c) {
    const float lv = l[c * NL + n];
    sk = fmaf(w_k[t * LC + c], lv, sk);
    sv = fmaf(w_v[t * LC + c], lv, sv);
  }
  klang[t * NL + n] = sk * mk * 0.0625f;   // fold K^-0.5 = 1/16 into k
  vlang[t * NL + n] = sv * mk;
  if (t == 0) simbias[n] = 10000.f * (mk - 1.f);
}

// --- main fused kernel: one batch per workgroup, 8 waves ----------------------
__global__ __launch_bounds__(256)
void fused_main(const float* __restrict__ x,    const float* __restrict__ PE,
                const float* __restrict__ w_lin,const float* __restrict__ b_lin,
                const float* __restrict__ b_vis,const float* __restrict__ b_q,
                const float* __restrict__ b_W,  const float* __restrict__ b_mm,
                const __bf16* __restrict__ wb_vis, const __bf16* __restrict__ wb_q,
                const __bf16* __restrict__ wb_W,   const __bf16* __restrict__ wb_mm,
                const float* __restrict__ kvlang,  // klang|vlang|simbias packed
                float* __restrict__ out)
{
  __shared__ __align__(16) __bf16 smA[MPAD * CDIM];   // x -> q_out -> lang -> ...
  __shared__ __align__(16) __bf16 smVis[MPAD * CDIM]; // vis (kept until product)
  __shared__ __align__(16) __bf16 smT[MPAD * CDIM];   // qpre -> attnOut -> product
  __shared__ __align__(16) float  smKV[KVCHUNKS * 4]; // klang | vlang | simbias

  const int b = blockIdx.x;
  const int t = threadIdx.x;        // channel 0..255 for per-channel stages

  // Kick off the async LDS staging of language K/V first: it drains only at
  // the attention stage, hiding behind stages 0..3b (two WMMA GEMMs).
#if HAVE_ASYNC_LDS
  {
    const char* gsrc = (const char*)kvlang;
    char*       ldst = (char*)smKV;
#pragma unroll 1
    for (int i = t; i < KVCHUNKS; i += 256) {
      __builtin_amdgcn_global_load_async_to_lds_b128(
          (gv4p)(gsrc + 16 * i), (lv4p)(ldst + 16 * i), 0, 0);
    }
  }
#else
  {
#pragma unroll 1
    for (int i = t; i < KVBYTES / 4; i += 256) smKV[i] = kvlang[i];
  }
#endif

  __builtin_prefetch(wb_vis + t * 64, 0, 0);   // warm weight panels into L2/WGP$
  __builtin_prefetch(wb_q   + t * 64, 0, 0);

  // Stage 0: x[b] (49x256 f32) -> bf16 LDS, zero-pad rows 49..63.
  // Non-temporal: x is a 205MB single-use stream, keep it out of L2.
  const float* xb = x + (size_t)b * HW * CDIM;
#pragma unroll 1
  for (int m = 0; m < HW; ++m)
    smA[m * CDIM + t] = (__bf16)__builtin_nontemporal_load(xb + m * CDIM + t);
#pragma unroll
  for (int m = HW; m < MPAD; ++m) smA[m * CDIM + t] = (__bf16)0.f;
  __syncthreads();

  // Stage 1: vis = gelu(x @ Wvis^T + b_vis)  -> smVis
  gemm_fused(smA, wb_vis, b_vis, 1, smVis, nullptr);
  __syncthreads();

  // Stage 2: qpre = x @ Wq^T + b_q  -> smT
  gemm_fused(smA, wb_q, b_q, 0, smT, nullptr);
  __syncthreads();

  // Stage 3: instance-norm over HW per channel, in place on smT (rows < 49)
  {
    float s1 = 0.f, s2 = 0.f;
#pragma unroll 1
    for (int m = 0; m < HW; ++m) {
      const float f = (float)smT[m * CDIM + t];
      s1 += f; s2 += f * f;
    }
    const float mu  = s1 * (1.f / HW);
    const float var = s2 * (1.f / HW) - mu * mu;
    const float rs  = rsqrtf(var + EPSV);
#pragma unroll 1
    for (int m = 0; m < HW; ++m)
      smT[m * CDIM + t] = (__bf16)(((float)smT[m * CDIM + t] - mu) * rs);
  }
  __syncthreads();

  // Stage 3b: concat PE, Linear(50->49) along HW: q_out -> smA (x dead now)
  {
    const float pe = PE[(size_t)b * CDIM + t];
#pragma unroll 1
    for (int m = 0; m < HW; ++m) {
      float s = b_lin[m];
#pragma unroll 1
      for (int j = 0; j < HW; ++j)
        s = fmaf(w_lin[m * 50 + j], (float)smT[j * CDIM + t], s);
      s = fmaf(w_lin[m * 50 + HW], pe, s);
      smA[m * CDIM + t] = (__bf16)s;
    }
  }
  // Drain the async K/V staging (per-wave), then block-wide barrier makes it
  // visible to everyone before attention reads smKV.
#if HAVE_ASYNC_LDS
#if __has_builtin(__builtin_amdgcn_s_wait_asynccnt)
  __builtin_amdgcn_s_wait_asynccnt(0);
#else
  asm volatile("s_wait_asynccnt 0x0" ::: "memory");
#endif
#endif
  __syncthreads();

  // Stage 4: 2-head attention against LDS-staged language K/V -> smT
  {
    const float* kl = smKV;                     // 256x20
    const float* vl = smKV + CDIM * NL;         // 256x20
    const float* sb = smKV + 2 * CDIM * NL;     // 20
    const int h = t >> 7;         // head
    const int m = t & 127;        // spatial row
    if (m < HW) {
      float sim[NL];
#pragma unroll
      for (int n = 0; n < NL; ++n) sim[n] = sb[n];
#pragma unroll 1
      for (int d = 0; d < KH; ++d) {
        const float qd = (float)smA[m * CDIM + h * KH + d];
        const float* kr = kl + (h * KH + d) * NL;
#pragma unroll
        for (int n = 0; n < NL; ++n) sim[n] = fmaf(qd, kr[n], sim[n]);
      }
      float mx = sim[0];
#pragma unroll
      for (int n = 1; n < NL; ++n) mx = fmaxf(mx, sim[n]);
      float se = 0.f;
#pragma unroll
      for (int n = 0; n < NL; ++n) { sim[n] = __expf(sim[n] - mx); se += sim[n]; }
      const float rs = 1.f / se;
#pragma unroll 1
      for (int e = 0; e < KH; ++e) {
        const float* vr = vl + (h * KH + e) * NL;
        float o = 0.f;
#pragma unroll
        for (int n = 0; n < NL; ++n) o = fmaf(sim[n], vr[n], o);
        smT[m * CDIM + h * KH + e] = (__bf16)(o * rs);
      }
    }
    // zero-pad rows 49..63 of smT (stale qpre values live there)
#pragma unroll
    for (int m2 = HW; m2 < MPAD; ++m2) smT[m2 * CDIM + t] = (__bf16)0.f;
  }
  __syncthreads();

  // Stage 5: lang_pre = attnOut @ Ww^T + b_W  -> smA
  gemm_fused(smT, wb_W, b_W, 0, smA, nullptr);
  __syncthreads();

  // Stage 6: instance-norm lang over HW per channel, in place on smA
  {
    float s1 = 0.f, s2 = 0.f;
#pragma unroll 1
    for (int m = 0; m < HW; ++m) {
      const float f = (float)smA[m * CDIM + t];
      s1 += f; s2 += f * f;
    }
    const float mu  = s1 * (1.f / HW);
    const float var = s2 * (1.f / HW) - mu * mu;
    const float rs  = rsqrtf(var + EPSV);
#pragma unroll 1
    for (int m = 0; m < HW; ++m)
      smA[m * CDIM + t] = (__bf16)(((float)smA[m * CDIM + t] - mu) * rs);
  }
  __syncthreads();

  // Stage 7: product vis * lang -> smT (rows 49..63 already zero)
#pragma unroll 1
  for (int m = 0; m < HW; ++m)
    smT[m * CDIM + t] = (__bf16)((float)smVis[m * CDIM + t] *
                                 (float)smA[m * CDIM + t]);
  __syncthreads();

  // Stage 8: mm = gelu(prod @ Wmm^T + b_mm) -> global out (B, HW, C) f32 (NT)
  gemm_fused(smT, wb_mm, b_mm, 1, nullptr, out + (size_t)b * HW * CDIM);
}

// ---------------------------------------------------------------------------
extern "C" void kernel_launch(void* const* d_in, const int* in_sizes, int n_in,
                              void* d_out, int out_size, void* d_ws, size_t ws_size,
                              hipStream_t stream) {
  const float* x      = (const float*)d_in[0];
  const float* l      = (const float*)d_in[1];
  const float* l_mask = (const float*)d_in[2];
  const float* PE     = (const float*)d_in[3];
  const float* w_vis  = (const float*)d_in[4];
  const float* b_vis  = (const float*)d_in[5];
  const float* w_q    = (const float*)d_in[6];
  const float* b_q    = (const float*)d_in[7];
  const float* w_k    = (const float*)d_in[8];
  const float* b_k    = (const float*)d_in[9];
  const float* w_v    = (const float*)d_in[10];
  const float* b_v    = (const float*)d_in[11];
  const float* w_W    = (const float*)d_in[12];
  const float* b_W    = (const float*)d_in[13];
  const float* w_lin  = (const float*)d_in[14];
  const float* b_lin  = (const float*)d_in[15];
  const float* w_mm   = (const float*)d_in[16];
  const float* b_mm   = (const float*)d_in[17];

  // workspace layout:
  //   [0)       4 x 65536 bf16 weight copies (vis, q, W, mm)   = 524288 B
  //   [524288)  klang  256x20 f32 (pre-scaled by 1/16, masked) =  20480 B
  //   [544768)  vlang  256x20 f32 (masked)                     =  20480 B
  //   [565248)  simbias 20 f32                                 =     80 B
  //   (klang|vlang|simbias contiguous -> one async-copy source block)
  char* ws = (char*)d_ws;
  __bf16* wb      = (__bf16*)ws;
  float*  klang   = (float*)(ws + 524288);
  float*  vlang   = (float*)(ws + 544768);
  float*  simbias = (float*)(ws + 565248);

  convert_weights<<<1024, 256, 0, stream>>>(w_vis, w_q, w_W, w_mm, wb);
  build_kv<<<NL, 256, 0, stream>>>(l, l_mask, w_k, b_k, w_v, b_v,
                                   klang, vlang, simbias);
  fused_main<<<4096, 256, 0, stream>>>(
      x, PE, w_lin, b_lin, b_vis, b_q, b_W, b_mm,
      wb + 0 * 65536, wb + 1 * 65536, wb + 2 * 65536, wb + 3 * 65536,
      klang, (float*)d_out);
}
